// GCN_83116207112815
// MI455X (gfx1250) — matrix operand order repaired and checked by reading, compiled, not run on hardware
//
#include <hip/hip_runtime.h>
#include <hip/hip_bf16.h>

// GCN on MI455X (gfx1250). Dominant cost: edge gather/scatter (~2.1 GB -> ~100us
// at 23.3 TB/s); GEMMs are trivial and run on v_wmma_f32_16x16x32_f16 with f16
// inputs staged in LDS and f32 accumulation.

typedef __attribute__((ext_vector_type(16))) _Float16 v16h;
typedef __attribute__((ext_vector_type(8)))  _Float16 v8h;
typedef __attribute__((ext_vector_type(8)))  float    v8f;

#define HID 128
#define NNODES 40000
#define NEDGES 640000
#define NGRAPHS 64

// ---------------------------------------------------------------- utilities
__global__ void gcn_zero_kernel(float* __restrict__ p, int n) {
    int i = blockIdx.x * blockDim.x + threadIdx.x;
    if (i < n) p[i] = 0.0f;
}

// deg[dst] += 1 per edge (self-loop +1 folded into dinv kernel)
__global__ void gcn_degree_kernel(const long long* __restrict__ ei,
                                  float* __restrict__ deg, int E) {
    int e = blockIdx.x * blockDim.x + threadIdx.x;
    if (e < E) {
        long long d = ei[(long long)E + e];
        __hip_atomic_fetch_add(&deg[d], 1.0f, __ATOMIC_RELAXED, __HIP_MEMORY_SCOPE_AGENT);
    }
}

// in-place: dinv[i] = rsqrt(max(deg[i] + 1, 1))   (+1 = self loop)
__global__ void gcn_dinv_kernel(float* __restrict__ dinv, int n) {
    int i = blockIdx.x * blockDim.x + threadIdx.x;
    if (i < n) dinv[i] = __frsqrt_rn(fmaxf(dinv[i] + 1.0f, 1.0f));
}

// ------------------------------------------------------------- WMMA GEMM
// T[M x 128] = A[M x K] * W[K x 128], K <= 128 (zero padded).
// Block: 256 threads = 8 waves; each wave owns a 16-row strip of a 128-row
// M-tile and all 8 N-tiles (8 x v8f accumulators). A and W^T staged in LDS
// as f16 (2 x 32 KB).
__global__ void __launch_bounds__(256)
gcn_gemm_kernel(const float* __restrict__ A, const float* __restrict__ W,
                float* __restrict__ T, int M, int K) {
    __shared__ _Float16 As[128][128];   // [row][k]
    __shared__ _Float16 Wt[128][128];   // transposed: [n][k]

    const int tid = threadIdx.x;
    const int m0  = blockIdx.x * 128;

    // Stage W^T (f32 -> f16), zero-pad K dimension.
    for (int i = tid; i < 128 * 128; i += 256) {
        int k = i >> 7, n = i & 127;
        float w = (k < K) ? W[k * HID + n] : 0.0f;
        Wt[n][k] = (_Float16)w;
    }
    // Stage A tile (f32 -> f16), zero-pad rows/K.
    for (int i = tid; i < 128 * 128; i += 256) {
        int r = i >> 7, k = i & 127;
        int row = m0 + r;
        float a = (row < M && k < K) ? A[(long long)row * K + k] : 0.0f;
        As[r][k] = (_Float16)a;
    }
    __syncthreads();

    const int wave = tid >> 5;        // 0..7 -> M strip
    const int lane = tid & 31;
    const int l15  = lane & 15;
    const int hi   = lane >> 4;       // half-wave select
    const int arow = wave * 16 + l15; // A row within tile for this lane
    const int asel = hi * 8;          // A K-interleave per ISA 7.12.2

    v8f acc[8];
    for (int nt = 0; nt < 8; ++nt) acc[nt] = (v8f){};

    const int ktiles = (K + 31) >> 5;
    for (int kt = 0; kt < ktiles; ++kt) {
        const int k0 = kt << 5;
        // A fragment 16x32 f16: lane<16 holds K=k0+0..15; lane>=16 K=k0+16..31.
        v8h a_lo = *(const v8h*)&As[arow][k0 + hi * 16];
        v8h a_hi = *(const v8h*)&As[arow][k0 + hi * 16 + 8];
        v16h afrag;
        #pragma unroll
        for (int e = 0; e < 8; ++e) { afrag[e] = a_lo[e]; afrag[e + 8] = a_hi[e]; }

        #pragma unroll
        for (int nt = 0; nt < 8; ++nt) {
            // B fragment 32x16: lane<16 -> col l15, K=k0..k0+15; lane>=16 -> K+16.
            const int col = nt * 16 + l15;
            v16h bfrag = *(const v16h*)&Wt[col][k0 + hi * 16];
            acc[nt] = __builtin_amdgcn_wmma_f32_16x16x32_f16(
                false, afrag, false, bfrag, (short)0, acc[nt], false, false);
        }
    }

    // C/D layout: lanes 0-15 N=lane, M=v; lanes 16-31 N=lane-16, M=v+8.
    const int rbase = m0 + wave * 16 + hi * 8;
    #pragma unroll
    for (int nt = 0; nt < 8; ++nt) {
        const int c = nt * 16 + l15;
        #pragma unroll
        for (int v = 0; v < 8; ++v) {
            int r = rbase + v;
            if (r < M) T[(long long)r * HID + c] = acc[nt][v];
        }
    }
}

// --------------------------------------------------------- edge scatter-add
// One wave (32 lanes) per edge; each lane moves 4 contiguous floats (b128
// gather, 4 f32 global atomics). T (20 MB) and O (20 MB) both fit in the
// 192 MB L2, so the random gather/scatter stays on-die.
__global__ void gcn_edge_kernel(const float* __restrict__ T,
                                const long long* __restrict__ ei,
                                const float* __restrict__ dinv,
                                float* __restrict__ O, int E) {
    int gid  = blockIdx.x * blockDim.x + threadIdx.x;
    int edge = gid >> 5;
    int lane = gid & 31;
    if (edge >= E) return;
    long long s = ei[edge];
    long long d = ei[(long long)E + edge];
    float nrm = dinv[s] * dinv[d];
    float4 v = ((const float4*)(T + s * HID))[lane];
    float* o = (float*)(O + d * HID) + lane * 4;
    __hip_atomic_fetch_add(o + 0, v.x * nrm, __ATOMIC_RELAXED, __HIP_MEMORY_SCOPE_AGENT);
    __hip_atomic_fetch_add(o + 1, v.y * nrm, __ATOMIC_RELAXED, __HIP_MEMORY_SCOPE_AGENT);
    __hip_atomic_fetch_add(o + 2, v.z * nrm, __ATOMIC_RELAXED, __HIP_MEMORY_SCOPE_AGENT);
    __hip_atomic_fetch_add(o + 3, v.w * nrm, __ATOMIC_RELAXED, __HIP_MEMORY_SCOPE_AGENT);
}

// H = act(O + T*dinv^2 + b)  — fused self-loop + bias + ReLU streaming pass.
__global__ void gcn_finalize_kernel(const float* __restrict__ O,
                                    const float* __restrict__ T,
                                    const float* __restrict__ dinv,
                                    const float* __restrict__ b,
                                    float* __restrict__ H, int N, int do_relu) {
    int gid = blockIdx.x * blockDim.x + threadIdx.x;
    if (gid >= N * HID) return;
    int i = gid >> 7, f = gid & 127;
    float di = dinv[i];
    float v = O[gid] + T[gid] * di * di + b[f];
    if (do_relu) v = fmaxf(v, 0.0f);
    H[gid] = v;
}

// ------------------------------------------------------------------ pooling
__global__ void gcn_pool_kernel(const float* __restrict__ H,
                                const long long* __restrict__ batch,
                                float* __restrict__ sums,
                                float* __restrict__ cnts, int N) {
    int gid  = blockIdx.x * blockDim.x + threadIdx.x;
    int node = gid >> 5;
    int lane = gid & 31;
    if (node >= N) return;
    long long g = batch[node];
    float4 v = ((const float4*)(H + (long long)node * HID))[lane];
    float* s = sums + g * HID + lane * 4;
    __hip_atomic_fetch_add(s + 0, v.x, __ATOMIC_RELAXED, __HIP_MEMORY_SCOPE_AGENT);
    __hip_atomic_fetch_add(s + 1, v.y, __ATOMIC_RELAXED, __HIP_MEMORY_SCOPE_AGENT);
    __hip_atomic_fetch_add(s + 2, v.z, __ATOMIC_RELAXED, __HIP_MEMORY_SCOPE_AGENT);
    __hip_atomic_fetch_add(s + 3, v.w, __ATOMIC_RELAXED, __HIP_MEMORY_SCOPE_AGENT);
    if (lane == 0)
        __hip_atomic_fetch_add(&cnts[g], 1.0f, __ATOMIC_RELAXED, __HIP_MEMORY_SCOPE_AGENT);
}

// out[g][c] = mean(pooled[g]) . Wl[:,c] + bl[c]   (64x2, trivial)
__global__ void gcn_head_kernel(const float* __restrict__ sums,
                                const float* __restrict__ cnts,
                                const float* __restrict__ Wl,
                                const float* __restrict__ bl,
                                float* __restrict__ out) {
    int gid = blockIdx.x * blockDim.x + threadIdx.x;
    if (gid >= NGRAPHS * 2) return;
    int g = gid >> 1, c = gid & 1;
    float inv = 1.0f / fmaxf(cnts[g], 1.0f);
    float acc = 0.0f;
    for (int k = 0; k < HID; ++k)
        acc += sums[g * HID + k] * Wl[k * 2 + c];
    out[gid] = acc * inv + bl[c];
}

// ------------------------------------------------------------------ driver
extern "C" void kernel_launch(void* const* d_in, const int* in_sizes, int n_in,
                              void* d_out, int out_size, void* d_ws, size_t ws_size,
                              hipStream_t stream) {
    const float*     x   = (const float*)d_in[0];
    const long long* ei  = (const long long*)d_in[1];
    const long long* bat = (const long long*)d_in[2];
    const float* W1 = (const float*)d_in[3];  const float* b1 = (const float*)d_in[4];
    const float* W2 = (const float*)d_in[5];  const float* b2 = (const float*)d_in[6];
    const float* W3 = (const float*)d_in[7];  const float* b3 = (const float*)d_in[8];
    const float* Wl = (const float*)d_in[9];  const float* bl = (const float*)d_in[10];
    float* out = (float*)d_out;

    const int N = NNODES, E = NEDGES;
    const long long NH = (long long)N * HID;

    // workspace layout (bytes, 256-aligned)
    char* ws = (char*)d_ws;
    size_t off = 0;
    auto carve = [&](size_t bytes) { char* p = ws + off; off = (off + bytes + 255) & ~(size_t)255; return p; };
    float* dinv = (float*)carve((size_t)N * 4);
    float* T    = (float*)carve((size_t)NH * 4);
    float* O    = (float*)carve((size_t)NH * 4);
    float* H    = (float*)carve((size_t)NH * 4);
    float* sums = (float*)carve((size_t)NGRAPHS * HID * 4 + NGRAPHS * 4); // sums + cnts
    float* cnts = sums + NGRAPHS * HID;
    (void)ws_size; (void)n_in; (void)in_sizes; (void)out_size;

    const int B = 256;
    const int mtiles = (N + 127) / 128;
    const int nh_grid = (int)((NH + B - 1) / B);
    const int edge_grid = (E * 32 + B - 1) / B;

    // degrees -> dinv
    gcn_zero_kernel<<<(N + B - 1) / B, B, 0, stream>>>(dinv, N);
    gcn_degree_kernel<<<(E + B - 1) / B, B, 0, stream>>>(ei, dinv, E);
    gcn_dinv_kernel<<<(N + B - 1) / B, B, 0, stream>>>(dinv, N);

    // layer 1 (K=7, ReLU)
    gcn_gemm_kernel<<<mtiles, B, 0, stream>>>(x, W1, T, N, 7);
    gcn_zero_kernel<<<nh_grid, B, 0, stream>>>(O, (int)NH);
    gcn_edge_kernel<<<edge_grid, B, 0, stream>>>(T, ei, dinv, O, E);
    gcn_finalize_kernel<<<nh_grid, B, 0, stream>>>(O, T, dinv, b1, H, N, 1);

    // layer 2 (K=128, ReLU)
    gcn_gemm_kernel<<<mtiles, B, 0, stream>>>(H, W2, T, N, HID);
    gcn_zero_kernel<<<nh_grid, B, 0, stream>>>(O, (int)NH);
    gcn_edge_kernel<<<edge_grid, B, 0, stream>>>(T, ei, dinv, O, E);
    gcn_finalize_kernel<<<nh_grid, B, 0, stream>>>(O, T, dinv, b2, H, N, 1);

    // layer 3 (K=128, no ReLU)
    gcn_gemm_kernel<<<mtiles, B, 0, stream>>>(H, W3, T, N, HID);
    gcn_zero_kernel<<<nh_grid, B, 0, stream>>>(O, (int)NH);
    gcn_edge_kernel<<<edge_grid, B, 0, stream>>>(T, ei, dinv, O, E);
    gcn_finalize_kernel<<<nh_grid, B, 0, stream>>>(O, T, dinv, b3, H, N, 0);

    // global mean pool + classifier head
    gcn_zero_kernel<<<(NGRAPHS * HID + NGRAPHS + B - 1) / B, B, 0, stream>>>(sums, NGRAPHS * HID + NGRAPHS);
    gcn_pool_kernel<<<(N * 32 + B - 1) / B, B, 0, stream>>>(H, bat, sums, cnts, N);
    gcn_head_kernel<<<1, 128, 0, stream>>>(sums, cnts, Wl, bl, out);
}
